// MessagePassingEdgeModule_44942537785402
// MI455X (gfx1250) — matrix operand order repaired and checked by reading, not compiled, not run on hardware
//
#include <hip/hip_runtime.h>
#include <hip/hip_bf16.h>
#include <stdint.h>

typedef __attribute__((ext_vector_type(16))) __bf16 v16bf;
typedef __attribute__((ext_vector_type(8)))  __bf16 v8bf;
typedef __attribute__((ext_vector_type(8)))  float  v8f;

#define DIM   128
#define IN_F  512
#define LPAD  136   // padded row stride in bf16 elements (bank-conflict mitigation)

// CDNA5 async global->LDS copy (ASYNCcnt-tracked), per CDNA5 ISA
#define ASYNC_LDS_B128(ldsOff, gAddr)                                      \
  asm volatile("global_load_async_to_lds_b128 %0, %1, off"                 \
               :: "v"(ldsOff), "v"(gAddr) : "memory")
#define WAIT_ASYNCCNT0() asm volatile("s_wait_asynccnt 0x0" ::: "memory")

// ---------------------------------------------------------------------------
// Prep: convert W1 [512x128] and W2 [128x128] from f32 to bf16 in workspace.
// ---------------------------------------------------------------------------
__global__ void convert_weights_kernel(const float* __restrict__ W1,
                                       const float* __restrict__ W2,
                                       __bf16* __restrict__ w1b,
                                       __bf16* __restrict__ w2b) {
  int i = blockIdx.x * blockDim.x + threadIdx.x;
  if (i < IN_F * DIM) w1b[i] = (__bf16)W1[i];
  if (i < DIM * DIM)  w2b[i] = (__bf16)W2[i];
}

// ---------------------------------------------------------------------------
// Main: per-edge 2-layer MLP via bf16 WMMA, f32 accumulate.
// Block = 256 threads = 8 waves. Each wave: 16 edges x 128 outputs.
// ---------------------------------------------------------------------------
__launch_bounds__(256, 2)
__global__ void edge_mlp_wmma_kernel(const float* __restrict__ x,
                                     const float* __restrict__ edge_attr,
                                     const float* __restrict__ u,
                                     const __bf16* __restrict__ w1b,
                                     const __bf16* __restrict__ w2b,
                                     const float* __restrict__ b1,
                                     const float* __restrict__ b2,
                                     const int* __restrict__ edge_index,
                                     const int* __restrict__ batch,
                                     float* __restrict__ out,
                                     int E) {
  __shared__ __bf16 wchunk[32][LPAD];     // staged 32x128 weight chunk (shared)
  __shared__ __bf16 hbuf[8][16][LPAD];    // per-wave hidden tile, bf16

  const int tid   = threadIdx.x;
  const int lane  = tid & 31;
  const int wv    = tid >> 5;
  const int lHalf = lane >> 4;            // 0: K-runs {0-7,16-23}; 1: {8-15,24-31}
  const int lLow  = lane & 15;            // edge row within tile / N column

  const long long tileBase = (long long)blockIdx.x * 128 + (long long)wv * 16;
  long long e = tileBase + lLow;
  if (e >= E) e = (long long)E - 1;       // clamp gather (stores are guarded)

  const int srcN = edge_index[e];
  const int dstN = edge_index[(long long)E + e];
  const int g    = batch[srcN];

  const float* pSrc = x         + (long long)srcN * DIM;
  const float* pDst = x         + (long long)dstN * DIM;
  const float* pEdg = edge_attr + e * DIM;
  const float* pU   = u         + (long long)g * DIM;

  // per-thread slice of the cooperative 8KB weight-chunk copy (2 x b128)
  const int cRow0 = tid >> 4;             // rows 0..15
  const int cQ0   = tid & 15;
  const unsigned ldsW0 = (unsigned)(uintptr_t)&wchunk[cRow0][cQ0 * 8];
  const unsigned ldsW1 = (unsigned)(uintptr_t)&wchunk[cRow0 + 16][cQ0 * 8];
  const int gOff0 = cRow0 * DIM + cQ0 * 8;
  const int gOff1 = (cRow0 + 16) * DIM + cQ0 * 8;

  // ---- layer-1 accumulators initialized with bias b1 ----
  v8f acc[8];
#pragma unroll
  for (int nt = 0; nt < 8; ++nt) {
    float bv = b1[nt * 16 + lLow];
#pragma unroll
    for (int r = 0; r < 8; ++r) acc[nt][r] = bv;
  }

  // ---- layer 1: K = 512 = 4 gathered sources x 4 chunks of 32 ----
#pragma unroll
  for (int seg = 0; seg < 4; ++seg) {
    const float* bp = (seg == 0) ? pSrc : (seg == 1) ? pDst
                     : (seg == 2) ? pEdg : pU;       // compile-time select
    for (int k4 = 0; k4 < 4; ++k4) {
      const int kc = seg * 4 + k4;
      __syncthreads();
      // async-stage 32x128 bf16 chunk of W1 into LDS (ASYNCcnt path)
      const __bf16* wsrc = w1b + kc * 32 * DIM;
      ASYNC_LDS_B128(ldsW0, (unsigned long long)(uintptr_t)(wsrc + gOff0));
      ASYNC_LDS_B128(ldsW1, (unsigned long long)(uintptr_t)(wsrc + gOff1));

      // A fragment (overlaps the async copy): 16-bit A 16x32 layout,
      // two 8-element K-runs per lane.
      const int off = k4 * 32 + lHalf * 8;
      float4 f0 = *(const float4*)(bp + off);
      float4 f1 = *(const float4*)(bp + off + 4);
      float4 f2 = *(const float4*)(bp + off + 16);
      float4 f3 = *(const float4*)(bp + off + 20);
      v16bf a;
      a[0]=(__bf16)f0.x;  a[1]=(__bf16)f0.y;  a[2]=(__bf16)f0.z;  a[3]=(__bf16)f0.w;
      a[4]=(__bf16)f1.x;  a[5]=(__bf16)f1.y;  a[6]=(__bf16)f1.z;  a[7]=(__bf16)f1.w;
      a[8]=(__bf16)f2.x;  a[9]=(__bf16)f2.y;  a[10]=(__bf16)f2.z; a[11]=(__bf16)f2.w;
      a[12]=(__bf16)f3.x; a[13]=(__bf16)f3.y; a[14]=(__bf16)f3.z; a[15]=(__bf16)f3.w;

      WAIT_ASYNCCNT0();
      __syncthreads();

      // B fragments preloaded in groups of 4 so WMMAs issue back-to-back
#pragma unroll
      for (int g4 = 0; g4 < 2; ++g4) {
        v16bf bf[4];
#pragma unroll
        for (int j = 0; j < 4; ++j) {
          const int nt = g4 * 4 + j;
#pragma unroll
          for (int s = 0; s < 16; ++s)
            bf[j][s] = wchunk[(s & 7) + lHalf * 8 + (s >> 3) * 16][nt * 16 + lLow];
        }
#pragma unroll
        for (int j = 0; j < 4; ++j) {
          const int nt = g4 * 4 + j;
          acc[nt] = __builtin_amdgcn_wmma_f32_16x16x32_bf16(
                      false, a, false, bf[j], (short)0, acc[nt], false, false);
        }
      }
    }
  }

  // ---- relu(h) -> LDS as bf16 (C layout: VGPR r => row r / r+8, N = lLow) ----
#pragma unroll
  for (int nt = 0; nt < 8; ++nt) {
#pragma unroll
    for (int r = 0; r < 8; ++r) {
      float v = acc[nt][r];
      v = v > 0.0f ? v : 0.0f;
      hbuf[wv][r + lHalf * 8][nt * 16 + lLow] = (__bf16)v;
    }
  }

  // ---- layer-2 accumulators initialized with bias b2 ----
  v8f acc2[8];
#pragma unroll
  for (int nt = 0; nt < 8; ++nt) {
    float bv = b2[nt * 16 + lLow];
#pragma unroll
    for (int r = 0; r < 8; ++r) acc2[nt][r] = bv;
  }

  // ---- layer 2: K = 128 in 4 chunks of 32 ----
  for (int kc = 0; kc < 4; ++kc) {
    __syncthreads();                       // orders h stores, protects wchunk reuse
    const __bf16* wsrc = w2b + kc * 32 * DIM;
    ASYNC_LDS_B128(ldsW0, (unsigned long long)(uintptr_t)(wsrc + gOff0));
    ASYNC_LDS_B128(ldsW1, (unsigned long long)(uintptr_t)(wsrc + gOff1));
    WAIT_ASYNCCNT0();
    __syncthreads();

    const int off0 = kc * 32 + lHalf * 8;  // run0; run1 = off0 + 16
    v8bf alo = *(const v8bf*)&hbuf[wv][lLow][off0];
    v8bf ahi = *(const v8bf*)&hbuf[wv][lLow][off0 + 16];
    v16bf a = __builtin_shufflevector(alo, ahi,
               0,1,2,3,4,5,6,7,8,9,10,11,12,13,14,15);

#pragma unroll
    for (int g4 = 0; g4 < 2; ++g4) {
      v16bf bf[4];
#pragma unroll
      for (int j = 0; j < 4; ++j) {
        const int nt = g4 * 4 + j;
#pragma unroll
        for (int s = 0; s < 16; ++s)
          bf[j][s] = wchunk[(s & 7) + lHalf * 8 + (s >> 3) * 16][nt * 16 + lLow];
      }
#pragma unroll
      for (int j = 0; j < 4; ++j) {
        const int nt = g4 * 4 + j;
        acc2[nt] = __builtin_amdgcn_wmma_f32_16x16x32_bf16(
                     false, a, false, bf[j], (short)0, acc2[nt], false, false);
      }
    }
  }

  // ---- final relu + coalesced f32 stores ----
#pragma unroll
  for (int nt = 0; nt < 8; ++nt) {
#pragma unroll
    for (int r = 0; r < 8; ++r) {
      long long er = tileBase + r + lHalf * 8;
      if (er < E) {
        float v = acc2[nt][r];
        out[er * DIM + nt * 16 + lLow] = v > 0.0f ? v : 0.0f;
      }
    }
  }
}

// ---------------------------------------------------------------------------
extern "C" void kernel_launch(void* const* d_in, const int* in_sizes, int n_in,
                              void* d_out, int out_size, void* d_ws, size_t ws_size,
                              hipStream_t stream) {
  const float* x         = (const float*)d_in[0];
  const float* edge_attr = (const float*)d_in[1];
  const float* u         = (const float*)d_in[2];
  const float* W1        = (const float*)d_in[3];
  const float* b1        = (const float*)d_in[4];
  const float* W2        = (const float*)d_in[5];
  const float* b2        = (const float*)d_in[6];
  const int*   edge_index= (const int*)d_in[7];
  const int*   batch     = (const int*)d_in[8];
  float*       out       = (float*)d_out;

  const int E = in_sizes[7] / 2;

  __bf16* w1b = (__bf16*)d_ws;
  __bf16* w2b = w1b + IN_F * DIM;

  convert_weights_kernel<<<(IN_F * DIM + 255) / 256, 256, 0, stream>>>(W1, W2, w1b, w2b);

  const int nBlocks = (E + 127) / 128;
  edge_mlp_wmma_kernel<<<nBlocks, 256, 0, stream>>>(
      x, edge_attr, u, w1b, w2b, b1, b2, edge_index, batch, out, E);
}
